// SAGEnorm_5806795784663
// MI455X (gfx1250) — matrix operand order repaired and checked by reading, not compiled, not run on hardware
//
#include <hip/hip_runtime.h>
#include <hip/hip_bf16.h>
#include <stdint.h>

// ---------------------------------------------------------------------------
// GraphSAGE (3-layer, mean aggregation) for MI455X / gfx1250.
//  - scatter: edge-parallel global_atomic_add_f32 (L2-resident features)
//  - gemm:    v_wmma_f32_16x16x32_bf16, fused [mean||h] @ [Wl;Wr] + bias,
//             double-buffered LDS tiles (ping-pong, 1 barrier / K-step)
//  - BN+ReLU fused elementwise pass; final 256-dot via wave32 shfl reduce
// ---------------------------------------------------------------------------

typedef __attribute__((ext_vector_type(16))) __bf16        v16bf;
typedef __attribute__((ext_vector_type(8)))  float         v8f;
typedef __attribute__((ext_vector_type(4)))  unsigned int  u32x4;
typedef __attribute__((ext_vector_type(4)))  float         f32x4;

#define NN      100000
#define NE      1600000
#define DIN     300
#define HID     256
#define K0PAD   608          // 19*32 >= 2*300
#define K1PAD   512          // 16*32 == 2*256
#define BN_EPS  1e-5f

union Frag { v16bf v; u32x4 q[2]; };

__device__ __forceinline__ unsigned short f2bf(float f) {
  unsigned int u = __float_as_uint(f);
  unsigned int r = (u + 0x7FFFu + ((u >> 16) & 1u)) >> 16;   // RNE
  return (unsigned short)r;
}

// -------------------------------- utility ----------------------------------
__global__ void zero_kernel(float* __restrict__ p, int n) {
  int i = blockIdx.x * blockDim.x + threadIdx.x;
  int stride = gridDim.x * blockDim.x;
  for (; i < n; i += stride) p[i] = 0.0f;
}

__global__ void deg_kernel(const long long* __restrict__ dst,
                           float* __restrict__ deg, int E) {
  int e = blockIdx.x * blockDim.x + threadIdx.x;
  if (e < E) unsafeAtomicAdd(&deg[(int)dst[e]], 1.0f);
}

// ------------------------------ edge scatter -------------------------------
// one thread per (edge, 4-feature chunk): msg[dst] += h[src]
__global__ void scatter_kernel(const float* __restrict__ h,
                               const long long* __restrict__ src,
                               const long long* __restrict__ dst,
                               float* __restrict__ msg, int E, int C, int D) {
  int idx = blockIdx.x * blockDim.x + threadIdx.x;
  if (idx >= E * C) return;
  int e = idx / C;
  int c = (idx - e * C) * 4;
  int s = (int)src[e];
  int d = (int)dst[e];
  f32x4 v = *(const f32x4*)(h + (size_t)s * D + c);
  float* o = msg + (size_t)d * D + c;
  unsafeAtomicAdd(o + 0, v.x);
  unsafeAtomicAdd(o + 1, v.y);
  unsafeAtomicAdd(o + 2, v.z);
  unsafeAtomicAdd(o + 3, v.w);
}

// ------------------------- bf16 operand preparation ------------------------
// A[n][k] = k<D ? msg[n][k]/max(deg,1) : (k<2D ? h[n][k-D] : 0)
__global__ void buildA_kernel(const float* __restrict__ msg,
                              const float* __restrict__ deg,
                              const float* __restrict__ h,
                              unsigned short* __restrict__ A,
                              int D, int Kpad) {
  int idx = blockIdx.x * blockDim.x + threadIdx.x;
  if (idx >= NN * Kpad) return;
  int n = idx / Kpad;
  int k = idx - n * Kpad;
  float val = 0.0f;
  if (k < D) {
    val = msg[(size_t)n * D + k] / fmaxf(deg[n], 1.0f);
  } else if (k < 2 * D) {
    val = h[(size_t)n * D + (k - D)];
  }
  A[idx] = f2bf(val);
}

// BT[n][k] = k<D ? Wl[k][n] : (k<2D ? Wr[k-D][n] : 0)   (K-major per column)
__global__ void buildBT_kernel(const float* __restrict__ Wl,
                               const float* __restrict__ Wr,
                               unsigned short* __restrict__ BT,
                               int D, int Kpad) {
  int idx = blockIdx.x * blockDim.x + threadIdx.x;
  if (idx >= HID * Kpad) return;
  int n = idx / Kpad;
  int k = idx - n * Kpad;
  float v = 0.0f;
  if (k < D)          v = Wl[(size_t)k * HID + n];
  else if (k < 2 * D) v = Wr[(size_t)(k - D) * HID + n];
  BT[idx] = f2bf(v);
}

// ------------------------------ WMMA GEMM ----------------------------------
// H[N x 256] = A[N x K](bf16) @ B[K x 256](bf16, pre-transposed BT) + bias
// block: 256 thr (8 waves), tile 128x128; wave tile 32x64 (2x4 wmma tiles)
// double-buffered LDS, one barrier per K-step
#define BM   128
#define BN   128
#define LDSP 40   // padded row stride in ushorts (80B, 16B-aligned)

__global__ __launch_bounds__(256)
void gemm_bf16_wmma(const unsigned short* __restrict__ A,
                    const unsigned short* __restrict__ BT,
                    const float* __restrict__ bias,
                    float* __restrict__ H, int Nrows, int K) {
  __shared__ __align__(16) unsigned short As[2][BM][LDSP];
  __shared__ __align__(16) unsigned short Bs[2][BN][LDSP];

  const int tid  = threadIdx.x;
  const int lane = tid & 31;
  const int wave = tid >> 5;
  const int wr = wave & 3;          // 4 row groups of 32
  const int wc = wave >> 2;         // 2 col groups of 64
  const int rowBase = blockIdx.y * BM;
  const int colBase = blockIdx.x * BN;
  const int laneh = lane >> 4;      // 0/1 half-wave select
  const int lanel = lane & 15;

  // staging assignment: thread t -> row t>>1, 16-u16 chunk (t&1)
  const int sr = tid >> 1;
  const int kq = (tid & 1) * 16;
  const bool aval = (rowBase + sr) < Nrows;
  const unsigned short* pa = A  + (size_t)(rowBase + sr) * K + kq;
  const unsigned short* pb = BT + (size_t)(colBase + sr) * K + kq;

  v8f acc[2][4];
#pragma unroll
  for (int m = 0; m < 2; ++m)
#pragma unroll
    for (int nb = 0; nb < 4; ++nb)
#pragma unroll
      for (int r = 0; r < 8; ++r) acc[m][nb][r] = 0.0f;

  const int nK = K >> 5;

  // ---- prologue: stage tile 0 into buffer 0 -------------------------------
  {
    u32x4 a0 = {0, 0, 0, 0}, a1 = {0, 0, 0, 0};
    if (aval) {
      a0 = ((const u32x4*)pa)[0];
      a1 = ((const u32x4*)pa)[1];
    }
    u32x4 b0 = ((const u32x4*)pb)[0];
    u32x4 b1 = ((const u32x4*)pb)[1];
    *(u32x4*)&As[0][sr][kq]     = a0;
    *(u32x4*)&As[0][sr][kq + 8] = a1;
    *(u32x4*)&Bs[0][sr][kq]     = b0;
    *(u32x4*)&Bs[0][sr][kq + 8] = b1;
  }

  for (int kt = 0; kt < nK; ++kt) {
    __syncthreads();                       // buffer kt&1 now visible
    const int cur = kt & 1;
    const int nxt = cur ^ 1;

    // ---- issue global loads for next tile (overlap with WMMA below) -------
    u32x4 a0 = {0, 0, 0, 0}, a1 = {0, 0, 0, 0};
    u32x4 b0 = {0, 0, 0, 0}, b1 = {0, 0, 0, 0};
    const bool more = (kt + 1) < nK;
    if (more) {
      const unsigned short* qa = pa + (size_t)(kt + 1) * 32;
      const unsigned short* qb = pb + (size_t)(kt + 1) * 32;
      if (aval) {
        a0 = ((const u32x4*)qa)[0];
        a1 = ((const u32x4*)qa)[1];
      }
      b0 = ((const u32x4*)qb)[0];
      b1 = ((const u32x4*)qb)[1];
    }

    // ---- fragments per documented CDNA5 16-bit WMMA lane layout -----------
    Frag af[2];
#pragma unroll
    for (int m = 0; m < 2; ++m) {
      int row = wr * 32 + m * 16 + lanel;                     // M = lane%16
      af[m].q[0] = *(const u32x4*)&As[cur][row][8 * laneh];        // K=e   +8h
      af[m].q[1] = *(const u32x4*)&As[cur][row][16 + 8 * laneh];   // K=e+8 +8h
    }
    Frag bf[4];
#pragma unroll
    for (int nb = 0; nb < 4; ++nb) {
      int n = wc * 64 + nb * 16 + lanel;                      // N = lane%16
      bf[nb].q[0] = *(const u32x4*)&Bs[cur][n][8 * laneh];         // K=e   +8h
      bf[nb].q[1] = *(const u32x4*)&Bs[cur][n][16 + 8 * laneh];
    }
#pragma unroll
    for (int m = 0; m < 2; ++m)
#pragma unroll
      for (int nb = 0; nb < 4; ++nb)
        acc[m][nb] = __builtin_amdgcn_wmma_f32_16x16x32_bf16(
            false, af[m].v, false, bf[nb].v, (short)0, acc[m][nb], false, false);

    // ---- commit next tile to the other buffer -----------------------------
    if (more) {
      *(u32x4*)&As[nxt][sr][kq]     = a0;
      *(u32x4*)&As[nxt][sr][kq + 8] = a1;
      *(u32x4*)&Bs[nxt][sr][kq]     = b0;
      *(u32x4*)&Bs[nxt][sr][kq + 8] = b1;
    }
  }

  // ---- epilogue: C layout VGPR r -> M = r + 8*(lane/16), N = lane%16 ------
#pragma unroll
  for (int m = 0; m < 2; ++m)
#pragma unroll
    for (int nb = 0; nb < 4; ++nb) {
      int gcol = colBase + wc * 64 + nb * 16 + lanel;
      float bsum = bias[gcol];
#pragma unroll
      for (int r = 0; r < 8; ++r) {
        int grow = rowBase + wr * 32 + m * 16 + r + 8 * laneh;
        if (grow < Nrows) H[(size_t)grow * HID + gcol] = acc[m][nb][r] + bsum;
      }
    }
}

// ------------------------- batchnorm column stats --------------------------
#define SROWS 512
__global__ void colstats_kernel(const float* __restrict__ H,
                                float* __restrict__ stats, int Nrows) {
  int c  = threadIdx.x;                 // 256 threads = 256 columns
  int r0 = blockIdx.x * SROWS;
  int r1 = r0 + SROWS; if (r1 > Nrows) r1 = Nrows;
  const float* p = H + (size_t)r0 * HID + c;
  float s = 0.0f, s2 = 0.0f;
  for (int r = r0; r < r1; ++r) {
    float v = *p;
    p += HID;
    s += v; s2 += v * v;
  }
  unsafeAtomicAdd(&stats[c], s);
  unsafeAtomicAdd(&stats[HID + c], s2);
}

__global__ void bn_relu_kernel(const float* __restrict__ H,
                               const float* __restrict__ stats,
                               const float* __restrict__ g,
                               const float* __restrict__ beta,
                               float* __restrict__ out, int Nrows) {
  int idx = blockIdx.x * blockDim.x + threadIdx.x;
  int total = Nrows * (HID / 4);
  if (idx >= total) return;
  int c = (idx & (HID / 4 - 1)) * 4;
  float invN = 1.0f / (float)Nrows;
  f32x4 v = *(const f32x4*)(H + (size_t)idx * 4);
  f32x4 o;
#pragma unroll
  for (int j = 0; j < 4; ++j) {
    float mu  = stats[c + j] * invN;
    float var = stats[HID + c + j] * invN - mu * mu;
    float sc  = g[c + j] * rsqrtf(var + BN_EPS);
    float sh  = beta[c + j] - mu * sc;
    float t   = v[j] * sc + sh;
    o[j] = t > 0.0f ? t : 0.0f;
  }
  *(f32x4*)(out + (size_t)idx * 4) = o;
}

// ------------------------ final projection [N,256]@[256,1] -----------------
__global__ void proj_kernel(const float* __restrict__ H,
                            const float* __restrict__ Wp,
                            const float* __restrict__ bp,
                            float* __restrict__ out, int Nrows) {
  int wave = threadIdx.x >> 5;
  int lane = threadIdx.x & 31;
  int node = blockIdx.x * 8 + wave;
  if (node >= Nrows) return;
  const float* p = H + (size_t)node * HID + lane;
  float s = 0.0f;
#pragma unroll
  for (int i = 0; i < 8; ++i) s += p[32 * i] * Wp[lane + 32 * i];
#pragma unroll
  for (int off = 16; off; off >>= 1) s += __shfl_xor(s, off, 32);
  if (lane == 0) out[node] = s + bp[0];
}

// ---------------------------------------------------------------------------
extern "C" void kernel_launch(void* const* d_in, const int* in_sizes, int n_in,
                              void* d_out, int out_size, void* d_ws, size_t ws_size,
                              hipStream_t stream) {
  const float*     x    = (const float*)d_in[0];
  const long long* ei   = (const long long*)d_in[1];
  const long long* esrc = ei;
  const long long* edst = ei + NE;
  const float* Wl0 = (const float*)d_in[2];
  const float* Wr0 = (const float*)d_in[3];
  const float* b0  = (const float*)d_in[4];
  const float* g0  = (const float*)d_in[5];
  const float* be0 = (const float*)d_in[6];
  const float* Wl1 = (const float*)d_in[7];
  const float* Wr1 = (const float*)d_in[8];
  const float* b1  = (const float*)d_in[9];
  const float* g1  = (const float*)d_in[10];
  const float* be1 = (const float*)d_in[11];
  const float* Wl2 = (const float*)d_in[12];
  const float* Wr2 = (const float*)d_in[13];
  const float* b2  = (const float*)d_in[14];
  const float* Wp  = (const float*)d_in[15];
  const float* bp  = (const float*)d_in[16];
  float* out = (float*)d_out;

  // workspace carve-up (256B aligned)
  char* ws = (char*)d_ws;
  size_t o = 0;
  float*          degb = (float*)(ws + o);          o += 400384;                 // N deg
  float*          msg  = (float*)(ws + o);          o += (size_t)NN * DIN * 4;   // 120 MB
  unsigned short* Abuf = (unsigned short*)(ws + o); o += (size_t)NN * K0PAD * 2; // 121.6 MB
  float*          Hbuf = (float*)(ws + o);          o += (size_t)NN * HID * 4;   // 102.4 MB
  float*          hbuf = (float*)(ws + o);          o += (size_t)NN * HID * 4;   // 102.4 MB
  unsigned short* Bbuf = (unsigned short*)(ws + o); o += (size_t)K0PAD * HID * 2;
  float*          stat = (float*)(ws + o);          o += 2048;

  const int TB = 256;
  dim3 gemmGrid(HID / BN, (NN + BM - 1) / BM);

  // degree (shared by all 3 layers)
  zero_kernel<<<(NN + TB - 1) / TB, TB, 0, stream>>>(degb, NN);
  deg_kernel<<<(NE + TB - 1) / TB, TB, 0, stream>>>(edst, degb, NE);

  // ---------------- layer 0: conv(300->256) + BN + ReLU ----------------
  zero_kernel<<<(NN * DIN + TB - 1) / TB, TB, 0, stream>>>(msg, NN * DIN);
  {
    int C = DIN / 4;
    long long work = (long long)NE * C;
    scatter_kernel<<<(int)((work + TB - 1) / TB), TB, 0, stream>>>(
        x, esrc, edst, msg, NE, C, DIN);
  }
  buildA_kernel<<<(NN * K0PAD + TB - 1) / TB, TB, 0, stream>>>(msg, degb, x, Abuf, DIN, K0PAD);
  buildBT_kernel<<<(K0PAD * HID + TB - 1) / TB, TB, 0, stream>>>(Wl0, Wr0, Bbuf, DIN, K0PAD);
  gemm_bf16_wmma<<<gemmGrid, TB, 0, stream>>>(Abuf, Bbuf, b0, Hbuf, NN, K0PAD);
  zero_kernel<<<2, TB, 0, stream>>>(stat, 2 * HID);
  colstats_kernel<<<(NN + SROWS - 1) / SROWS, HID, 0, stream>>>(Hbuf, stat, NN);
  bn_relu_kernel<<<(NN * (HID / 4) + TB - 1) / TB, TB, 0, stream>>>(Hbuf, stat, g0, be0, hbuf, NN);

  // ---------------- layer 1: conv(256->256) + BN + ReLU ----------------
  zero_kernel<<<(NN * HID + TB - 1) / TB, TB, 0, stream>>>(msg, NN * HID);
  {
    int C = HID / 4;
    long long work = (long long)NE * C;
    scatter_kernel<<<(int)((work + TB - 1) / TB), TB, 0, stream>>>(
        hbuf, esrc, edst, msg, NE, C, HID);
  }
  buildA_kernel<<<(NN * K1PAD + TB - 1) / TB, TB, 0, stream>>>(msg, degb, hbuf, Abuf, HID, K1PAD);
  buildBT_kernel<<<(K1PAD * HID + TB - 1) / TB, TB, 0, stream>>>(Wl1, Wr1, Bbuf, HID, K1PAD);
  gemm_bf16_wmma<<<gemmGrid, TB, 0, stream>>>(Abuf, Bbuf, b1, Hbuf, NN, K1PAD);
  zero_kernel<<<2, TB, 0, stream>>>(stat, 2 * HID);
  colstats_kernel<<<(NN + SROWS - 1) / SROWS, HID, 0, stream>>>(Hbuf, stat, NN);
  bn_relu_kernel<<<(NN * (HID / 4) + TB - 1) / TB, TB, 0, stream>>>(Hbuf, stat, g1, be1, hbuf, NN);

  // ---------------- layer 2: conv(256->256), then 256->1 proj ----------
  zero_kernel<<<(NN * HID + TB - 1) / TB, TB, 0, stream>>>(msg, NN * HID);
  {
    int C = HID / 4;
    long long work = (long long)NE * C;
    scatter_kernel<<<(int)((work + TB - 1) / TB), TB, 0, stream>>>(
        hbuf, esrc, edst, msg, NE, C, HID);
  }
  buildA_kernel<<<(NN * K1PAD + TB - 1) / TB, TB, 0, stream>>>(msg, degb, hbuf, Abuf, HID, K1PAD);
  buildBT_kernel<<<(K1PAD * HID + TB - 1) / TB, TB, 0, stream>>>(Wl2, Wr2, Bbuf, HID, K1PAD);
  gemm_bf16_wmma<<<gemmGrid, TB, 0, stream>>>(Abuf, Bbuf, b2, Hbuf, NN, K1PAD);

  proj_kernel<<<(NN + 7) / 8, TB, 0, stream>>>(Hbuf, Wp, bp, out, NN);

  (void)in_sizes; (void)n_in; (void)out_size; (void)ws_size;
}
